// SimpleLure_89215060672645
// MI455X (gfx1250) — matrix operand order repaired and checked by reading, not compiled, not run on hardware
//
#include <hip/hip_runtime.h>

// ---------------------------------------------------------------------------
// Lure-system scan for MI455X (gfx1250).
//   Persistent single-workgroup kernel: state + hot weights in 320KB LDS,
//   all GEMMs via v_wmma_f32_16x16x32_bf16 (wave32), cold weights streamed
//   from L2 as pre-converted bf16. Register-lean schedule (no spills): the
//   wave is VGPR-bound, so fragments are loaded just-in-time and only the
//   d_{t+1} tile is prefetched across the compute phase.
// ---------------------------------------------------------------------------

#define BATCH 64
#define NSTEP 1024
#define NDIM  64
#define NEDIM 64
#define NXDIM 256
#define NWDIM 128

typedef __attribute__((ext_vector_type(16))) __bf16 v16bf;
typedef __attribute__((ext_vector_type(8)))  float  v8f;

union Frag { uint4 u[2]; v16bf v; };

// LDS geometry (bytes). One 896B row per batch element: [x:512 | d:128 | w:256]
#define XDW_STRIDE 896u
#define X_BASE     0u
#define D_BASE     512u
#define W_BASE     640u
// Weight panels (bf16, row-major, XOR-swizzled rows)
#define A_LDS_OFF  57344u                      // 256 rows x 512B
#define C2_LDS_OFF (57344u + 131072u)          // 128 rows x 512B
#define B2_LDS_OFF (57344u + 131072u + 65536u) // 256 rows x 256B
#define LDS_BYTES  (57344 + 131072 + 65536 + 65536) // 319488 <= 320KB

__device__ __forceinline__ unsigned swz(unsigned off, int row, int mask) {
  return off ^ (unsigned)((row & mask) << 4);
}

__device__ __forceinline__ unsigned pack2bf(float lo, float hi) {
  union { __bf16 b[2]; unsigned u; } p;
  p.b[0] = (__bf16)lo;
  p.b[1] = (__bf16)hi;
  return p.u;
}

__device__ __forceinline__ float fast_tanh(float x) {
  float e = __expf(-2.0f * fabsf(x));  // v_exp_f32 path; stable for |x| large
  float r = (1.0f - e) / (1.0f + e);
  return copysignf(r, x);
}

// D = A x B + C, bf16 in / f32 accumulate
__device__ __forceinline__ v8f mm(v16bf a, v16bf b, v8f c) {
  return __builtin_amdgcn_wmma_f32_16x16x32_bf16(false, a, false, b,
                                                 (short)0, c, false, false);
}

// A-operand (X tile, rows = batch) from the swizzled state panel.
__device__ __forceinline__ v16bf afrag(const char* lds, unsigned regionBase,
                                       int mask, int m0, int k0, int r, int h) {
  int row = m0 + r;
  unsigned base = (unsigned)row * XDW_STRIDE + regionBase;
  Frag f;
  f.u[0] = *(const uint4*)(lds + base + swz((unsigned)(k0 + h * 32), row, mask));
  f.u[1] = *(const uint4*)(lds + base + swz((unsigned)(k0 + h * 32 + 16), row, mask));
  return f.v;
}

// B-operand (weight tile for Y = X * W^T) from an LDS weight panel.
__device__ __forceinline__ v16bf bfrag_lds(const char* lds, unsigned off,
                                           unsigned stride, int n0, int k0,
                                           int r, int h) {
  int row = n0 + r;
  unsigned base = off + (unsigned)row * stride;
  Frag f;
  f.u[0] = *(const uint4*)(lds + base + swz((unsigned)(k0 + h * 32), row, 15));
  f.u[1] = *(const uint4*)(lds + base + swz((unsigned)(k0 + h * 32 + 16), row, 15));
  return f.v;
}

// B-operand streamed from global (bf16, unswizzled row-major).
__device__ __forceinline__ v16bf bfrag_g(const unsigned short* w,
                                         unsigned rowBytes, int n0, int k0,
                                         int r, int h) {
  const char* p = (const char*)w + (size_t)(n0 + r) * rowBytes + k0 + h * 32;
  Frag f;
  f.u[0] = *(const uint4*)p;
  f.u[1] = *(const uint4*)(p + 16);
  return f.v;
}

__device__ __forceinline__ void fill_weight(char* lds, unsigned off,
                                            unsigned stride, const float* src,
                                            int rows, int cols, int tid, int nt) {
  int total = rows * cols / 2;
  for (int i = tid; i < total; i += nt) {
    int e = i * 2;
    int row = e / cols, k = e % cols;
    *(unsigned*)(lds + off + (unsigned)row * stride + swz((unsigned)(k * 2), row, 15)) =
        pack2bf(src[e], src[e + 1]);
  }
}

__device__ __forceinline__ void load_d_step(char* lds, const float* d, int t,
                                            int tid, int nt) {
  for (int i = tid; i < (BATCH * NDIM) / 4; i += nt) {
    int e = i * 4;
    int b = e >> 6, k = e & 63;
    float4 v = *(const float4*)(d + ((size_t)b * NSTEP + t) * NDIM + k);
    uint2 pk;
    pk.x = pack2bf(v.x, v.y);
    pk.y = pack2bf(v.z, v.w);
    *(uint2*)(lds + (unsigned)b * XDW_STRIDE + D_BASE + swz((unsigned)(k * 2), b, 7)) = pk;
  }
}

__global__ void f32_to_bf16_kernel(const float* __restrict__ src,
                                   __bf16* __restrict__ dst, int n) {
  int i = blockIdx.x * blockDim.x + threadIdx.x;
  if (i < n) dst[i] = (__bf16)src[i];
}

__global__ __launch_bounds__(512, 1)
void lure_scan_kernel(const float* __restrict__ dsig,
                      const float* __restrict__ Af,
                      const float* __restrict__ B2f,
                      const float* __restrict__ C2f,
                      const unsigned short* __restrict__ Bbf,
                      const unsigned short* __restrict__ D21bf,
                      const unsigned short* __restrict__ Cbf,
                      const unsigned short* __restrict__ Dbf,
                      const unsigned short* __restrict__ D12bf,
                      float* __restrict__ out) {
  __shared__ char lds[LDS_BYTES];
  const int tid  = threadIdx.x;
  const int lane = tid & 31;
  const int wv   = tid >> 5;       // 16 waves
  const int r    = lane & 15;
  const int h    = lane >> 4;

  // Warm the L2-streamed weight panels (contiguous 104KB starting at Bbf).
  for (int i = tid; i < (53248 * 2) / 64; i += 512)
    __builtin_prefetch((const char*)Bbf + i * 64, 0, 3);

  // ---- one-time init: weights -> LDS (bf16, swizzled), x0 = 0, stage d_0 ----
  fill_weight(lds, A_LDS_OFF,  512u, Af,  NXDIM, NXDIM, tid, 512);
  fill_weight(lds, C2_LDS_OFF, 512u, C2f, NWDIM, NXDIM, tid, 512);
  fill_weight(lds, B2_LDS_OFF, 256u, B2f, NXDIM, NWDIM, tid, 512);
  for (int i = tid; i < BATCH * 512 / 16; i += 512) {
    int row = i >> 5, c = i & 31;
    uint4 z; z.x = z.y = z.z = z.w = 0u;
    *(uint4*)(lds + (unsigned)row * XDW_STRIDE + (unsigned)(c * 16)) = z;
  }
  load_d_step(lds, dsig, 0, tid, 512);
  __syncthreads();

  const int m0 = (wv & 3) << 4;     // batch tile: 4 tiles of 16 rows
  const int q  = wv >> 2;           // 0..3

  #pragma unroll 1
  for (int t = 0; t < NSTEP; ++t) {
    // ---------- phase 1: z = x*C2^T + d*D21^T ; w = tanh(z) ----------
    {
      const int n0a = q << 5, n0b = n0a + 16;   // two of 8 z-column tiles
      v8f za = {0, 0, 0, 0, 0, 0, 0, 0};
      v8f zb = {0, 0, 0, 0, 0, 0, 0, 0};
      #pragma unroll
      for (int k0 = 0; k0 < 512; k0 += 64) {    // K-segment: x (C2, LDS)
        v16bf a = afrag(lds, X_BASE, 15, m0, k0, r, h);
        za = mm(a, bfrag_lds(lds, C2_LDS_OFF, 512u, n0a, k0, r, h), za);
        zb = mm(a, bfrag_lds(lds, C2_LDS_OFF, 512u, n0b, k0, r, h), zb);
      }
      #pragma unroll
      for (int k0 = 0; k0 < 128; k0 += 64) {    // K-segment: d (D21, L2)
        v16bf a = afrag(lds, D_BASE, 7, m0, k0, r, h);
        za = mm(a, bfrag_g(D21bf, 128u, n0a, k0, r, h), za);
        zb = mm(a, bfrag_g(D21bf, 128u, n0b, k0, r, h), zb);
      }
      #pragma unroll
      for (int i = 0; i < 8; ++i) {
        int row = m0 + i + (h << 3);
        unsigned rb = (unsigned)row * XDW_STRIDE + W_BASE;
        *(__bf16*)(lds + rb + swz((unsigned)((n0a + r) * 2), row, 15)) =
            (__bf16)fast_tanh(za[i]);
        *(__bf16*)(lds + rb + swz((unsigned)((n0b + r) * 2), row, 15)) =
            (__bf16)fast_tanh(zb[i]);
      }
    }
    __syncthreads();

    // ---------- phase 2: e (-> global) and x_next (-> regs) ----------
    // Prefetch d_{t+1} into registers first: its latency hides behind WMMAs.
    const int tn = (t + 1 < NSTEP) ? (t + 1) : t;
    const int e0 = tid * 4;              // chunk tid      (b = tid>>4)
    const int e1 = e0 + 2048;            // chunk tid+512  (b += 32)
    float4 dpre0 = *(const float4*)(dsig + ((size_t)(e0 >> 6) * NSTEP + tn) * NDIM + (e0 & 63));
    float4 dpre1 = *(const float4*)(dsig + ((size_t)(e1 >> 6) * NSTEP + tn) * NDIM + (e1 & 63));

    v8f xa[4];
    #pragma unroll
    for (int j = 0; j < 4; ++j) xa[j] = (v8f){0, 0, 0, 0, 0, 0, 0, 0};
    v8f ev = {0, 0, 0, 0, 0, 0, 0, 0};
    const int xn0 = q << 6;        // four x-column tiles per wave
    const int en0 = q << 4;        // one e-column tile per wave

    #pragma unroll
    for (int k0 = 0; k0 < 512; k0 += 64) {      // K-segment: x (A in LDS, C in L2)
      v16bf a = afrag(lds, X_BASE, 15, m0, k0, r, h);
      #pragma unroll
      for (int j = 0; j < 4; ++j)
        xa[j] = mm(a, bfrag_lds(lds, A_LDS_OFF, 512u, xn0 + j * 16, k0, r, h), xa[j]);
      ev = mm(a, bfrag_g(Cbf, 512u, en0, k0, r, h), ev);
    }
    #pragma unroll
    for (int k0 = 0; k0 < 128; k0 += 64) {      // K-segment: d (B, D in L2)
      v16bf a = afrag(lds, D_BASE, 7, m0, k0, r, h);
      #pragma unroll
      for (int j = 0; j < 4; ++j)
        xa[j] = mm(a, bfrag_g(Bbf, 128u, xn0 + j * 16, k0, r, h), xa[j]);
      ev = mm(a, bfrag_g(Dbf, 128u, en0, k0, r, h), ev);
    }
    #pragma unroll
    for (int k0 = 0; k0 < 256; k0 += 64) {      // K-segment: w (B2 in LDS, D12 in L2)
      v16bf a = afrag(lds, W_BASE, 15, m0, k0, r, h);
      #pragma unroll
      for (int j = 0; j < 4; ++j)
        xa[j] = mm(a, bfrag_lds(lds, B2_LDS_OFF, 256u, xn0 + j * 16, k0, r, h), xa[j]);
      ev = mm(a, bfrag_g(D12bf, 256u, en0, k0, r, h), ev);
    }
    #pragma unroll
    for (int i = 0; i < 8; ++i) {               // e[b, t, n] (f32 output)
      int b = m0 + i + (h << 3);
      out[((size_t)b * NSTEP + t) * NEDIM + en0 + r] = ev[i];
    }
    __syncthreads();

    // ---------- phase 3: commit x_next (bf16), commit staged d_{t+1} ----------
    #pragma unroll
    for (int j = 0; j < 4; ++j) {
      #pragma unroll
      for (int i = 0; i < 8; ++i) {
        int row = m0 + i + (h << 3);
        int col = xn0 + j * 16 + r;
        *(__bf16*)(lds + (unsigned)row * XDW_STRIDE +
                   swz((unsigned)(col * 2), row, 15)) = (__bf16)xa[j][i];
      }
    }
    {
      int b0i = e0 >> 6, k0i = e0 & 63;
      uint2 p0; p0.x = pack2bf(dpre0.x, dpre0.y); p0.y = pack2bf(dpre0.z, dpre0.w);
      *(uint2*)(lds + (unsigned)b0i * XDW_STRIDE + D_BASE + swz((unsigned)(k0i * 2), b0i, 7)) = p0;
      int b1i = e1 >> 6, k1i = e1 & 63;
      uint2 p1; p1.x = pack2bf(dpre1.x, dpre1.y); p1.y = pack2bf(dpre1.z, dpre1.w);
      *(uint2*)(lds + (unsigned)b1i * XDW_STRIDE + D_BASE + swz((unsigned)(k1i * 2), b1i, 7)) = p1;
    }
    __syncthreads();
  }
}

extern "C" void kernel_launch(void* const* d_in, const int* in_sizes, int n_in,
                              void* d_out, int out_size, void* d_ws, size_t ws_size,
                              hipStream_t stream) {
  (void)in_sizes; (void)n_in; (void)out_size; (void)ws_size;
  const float* dsig = (const float*)d_in[0];
  const float* A    = (const float*)d_in[1];
  const float* B    = (const float*)d_in[2];
  const float* B2   = (const float*)d_in[3];
  const float* C    = (const float*)d_in[4];
  const float* D    = (const float*)d_in[5];
  const float* D12  = (const float*)d_in[6];
  const float* C2   = (const float*)d_in[7];
  const float* D21  = (const float*)d_in[8];
  // d_in[9] = D22 (zeros; the reference's feedback term vanishes)

  // Workspace layout: bf16 copies of the L2-streamed (cold) weights.
  unsigned short* ws    = (unsigned short*)d_ws;
  unsigned short* Bbf   = ws;                 // 256*64
  unsigned short* D21bf = ws + 16384;         // 128*64
  unsigned short* Cbf   = ws + 24576;         // 64*256
  unsigned short* Dbf   = ws + 40960;         // 64*64
  unsigned short* D12bf = ws + 45056;         // 64*128  (total ~104 KB)

  f32_to_bf16_kernel<<<(16384 + 255) / 256, 256, 0, stream>>>(B,   (__bf16*)Bbf,   16384);
  f32_to_bf16_kernel<<<(8192  + 255) / 256, 256, 0, stream>>>(D21, (__bf16*)D21bf, 8192);
  f32_to_bf16_kernel<<<(16384 + 255) / 256, 256, 0, stream>>>(C,   (__bf16*)Cbf,   16384);
  f32_to_bf16_kernel<<<(4096  + 255) / 256, 256, 0, stream>>>(D,   (__bf16*)Dbf,   4096);
  f32_to_bf16_kernel<<<(8192  + 255) / 256, 256, 0, stream>>>(D12, (__bf16*)D12bf, 8192);

  lure_scan_kernel<<<1, 512, 0, stream>>>(dsig, A, B2, C2,
                                          Bbf, D21bf, Cbf, Dbf, D12bf,
                                          (float*)d_out);
}
